// Block_79164837200171
// MI455X (gfx1250) — matrix operand order repaired and checked by reading, not compiled, not run on hardware
//
#include <hip/hip_runtime.h>
#include <hip/hip_bf16.h>
#include <math.h>

typedef __bf16 bf16_t;
typedef __attribute__((ext_vector_type(16))) __bf16 v16bf;
typedef __attribute__((ext_vector_type(8)))  float  v8f;

#define DIM      768
#define HEADS    12
#define HEAD_DIM 64
#define HIDDEN   3072
#define BATCH    4
#define SEQ      1024
#define ROWS     (BATCH * SEQ)     // 4096
#define BH       (BATCH * HEADS)   // 48
#define ATT_SCALE 0.125f           // 64^-0.5
#define LN_EPS   1e-5f

enum { EPI_PLAIN = 0, EPI_SCALE = 1, EPI_BIAS_RESID = 2, EPI_BIAS_GELU_BF16 = 3 };

__device__ __forceinline__ float gelu_exact(float v) {
    return 0.5f * v * (1.0f + erff(v * 0.70710678118654752f));
}

// A fragment: 16x32 (MxK). Lane l<16 -> row M=l, K = {k0+0..7, k0+16..23};
// lanes 16..31 -> row M=l-16, K = {k0+8..15, k0+24..31}. p points at (row, k0+half*8).
template <typename T>
__device__ __forceinline__ v16bf load_a_frag(const T* __restrict__ p) {
    v16bf r;
#pragma unroll
    for (int i = 0; i < 8; ++i) {
        r[i]     = (bf16_t)p[i];
        r[i + 8] = (bf16_t)p[16 + i];
    }
    return r;
}

// B fragment: 32x16 (KxN), loaded from B^T (row-major [N,K]).
// Lane l<16 -> col N=l, K = {k0+0..7, k0+16..23}; lanes 16..31 -> col l-16, K = {k0+8..15, k0+24..31}.
// p points at (col, k0 + half*8).
__device__ __forceinline__ v16bf load_b_frag(const bf16_t* __restrict__ p) {
    v16bf r;
#pragma unroll
    for (int i = 0; i < 8; ++i) { r[i] = (bf16_t)p[i]; r[i + 8] = (bf16_t)p[16 + i]; }
    return r;
}

__device__ __forceinline__ void mma_step(v8f (&acc)[2][4],
                                         v16bf a0, v16bf a1,
                                         v16bf b0, v16bf b1, v16bf b2, v16bf b3) {
    acc[0][0] = __builtin_amdgcn_wmma_f32_16x16x32_bf16(false, a0, false, b0, (short)0, acc[0][0], false, false);
    acc[0][1] = __builtin_amdgcn_wmma_f32_16x16x32_bf16(false, a0, false, b1, (short)0, acc[0][1], false, false);
    acc[0][2] = __builtin_amdgcn_wmma_f32_16x16x32_bf16(false, a0, false, b2, (short)0, acc[0][2], false, false);
    acc[0][3] = __builtin_amdgcn_wmma_f32_16x16x32_bf16(false, a0, false, b3, (short)0, acc[0][3], false, false);
    acc[1][0] = __builtin_amdgcn_wmma_f32_16x16x32_bf16(false, a1, false, b0, (short)0, acc[1][0], false, false);
    acc[1][1] = __builtin_amdgcn_wmma_f32_16x16x32_bf16(false, a1, false, b1, (short)0, acc[1][1], false, false);
    acc[1][2] = __builtin_amdgcn_wmma_f32_16x16x32_bf16(false, a1, false, b2, (short)0, acc[1][2], false, false);
    acc[1][3] = __builtin_amdgcn_wmma_f32_16x16x32_bf16(false, a1, false, b3, (short)0, acc[1][3], false, false);
}

// C = A[M,K] x B[K,N] (B given transposed, bf16), fp32 accumulate via WMMA.
// Block = 128 threads = 4 waves stacked vertically; block tile 128x64;
// each wave computes 32 rows x 64 cols (2x4 accumulators), software-pipelined
// (next K-slice loads issued before current slice's 8 WMMAs).
// Requires: M % 128 == 0, N % 64 == 0, K % 32 == 0.
template <int EPI, typename TA>
__global__ __launch_bounds__(128) void gemm_wmma_kernel(
    const TA* __restrict__ A, const bf16_t* __restrict__ Bt,
    float* __restrict__ Cf, bf16_t* __restrict__ Cb,
    const float* __restrict__ bias, const float* __restrict__ resid,
    int M, int N, int K, int lda, int ldbt, int ldc,
    long long aBatch, long long btBatch, long long cBatch, float scale)
{
    const int z = blockIdx.z;
    const TA*     Ab  = A  + (long long)z * aBatch;
    const bf16_t* Btb = Bt + (long long)z * btBatch;
    const long long cOff = (long long)z * cBatch;

    const int wave = threadIdx.x >> 5;
    const int lane = threadIdx.x & 31;
    const int half = lane >> 4;
    const int l16  = lane & 15;

    const int m0 = blockIdx.y * 128 + wave * 32;
    const int n0 = blockIdx.x * 64;

    v8f acc[2][4] = {};

    const TA*     ap  = Ab  + (long long)(m0 + l16) * lda  + half * 8;
    const TA*     ap1 = ap  + (long long)16 * lda;
    const bf16_t* bp  = Btb + (long long)(n0 + l16) * ldbt + half * 8;
    const long long bstep = (long long)16 * ldbt;

    // Prologue: first K-slice
    v16bf a0 = load_a_frag(ap);
    v16bf a1 = load_a_frag(ap1);
    v16bf b0 = load_b_frag(bp);
    v16bf b1 = load_b_frag(bp + bstep);
    v16bf b2 = load_b_frag(bp + 2 * bstep);
    v16bf b3 = load_b_frag(bp + 3 * bstep);

    // Pipelined main loop: issue next slice's loads, then MMA current slice.
    for (int k0 = 32; k0 < K; k0 += 32) {
        __builtin_prefetch(ap + k0 + 32, 0, 1);
        __builtin_prefetch(bp + k0 + 32, 0, 1);
        v16bf na0 = load_a_frag(ap  + k0);
        v16bf na1 = load_a_frag(ap1 + k0);
        v16bf nb0 = load_b_frag(bp + k0);
        v16bf nb1 = load_b_frag(bp + bstep + k0);
        v16bf nb2 = load_b_frag(bp + 2 * bstep + k0);
        v16bf nb3 = load_b_frag(bp + 3 * bstep + k0);

        mma_step(acc, a0, a1, b0, b1, b2, b3);

        a0 = na0; a1 = na1;
        b0 = nb0; b1 = nb1; b2 = nb2; b3 = nb3;
    }
    // Epilogue K-slice
    mma_step(acc, a0, a1, b0, b1, b2, b3);

    // C layout: VGPR i -> row m0 + r*16 + half*8 + i, col n0 + 16*j + l16.
#pragma unroll
    for (int r = 0; r < 2; ++r) {
#pragma unroll
        for (int j = 0; j < 4; ++j) {
            const int col = n0 + 16 * j + l16;
#pragma unroll
            for (int i = 0; i < 8; ++i) {
                const int row = m0 + r * 16 + half * 8 + i;
                float v = acc[r][j][i];
                if (EPI == EPI_SCALE)          v *= scale;
                if (EPI == EPI_BIAS_RESID)     v += bias[col] + resid[(long long)row * ldc + col];
                if (EPI == EPI_BIAS_GELU_BF16) v = gelu_exact(v + bias[col]);
                const long long ci = cOff + (long long)row * ldc + col;
                if (EPI == EPI_BIAS_GELU_BF16) Cb[ci] = (bf16_t)v;
                else                           Cf[ci] = v;
            }
        }
    }
}

// LayerNorm over rows of [ROWS, DIM], bf16 output.
__global__ __launch_bounds__(256) void ln_kernel(
    const float* __restrict__ x, const float* __restrict__ g,
    const float* __restrict__ b, bf16_t* __restrict__ out)
{
    __shared__ float s1[256], s2[256];
    const int row = blockIdx.x;
    const float* xr = x + (long long)row * DIM;
    float sum = 0.f, sq = 0.f;
    for (int c = threadIdx.x; c < DIM; c += 256) {
        float v = xr[c]; sum += v; sq += v * v;
    }
    s1[threadIdx.x] = sum; s2[threadIdx.x] = sq;
    __syncthreads();
    for (int s = 128; s > 0; s >>= 1) {
        if (threadIdx.x < s) {
            s1[threadIdx.x] += s1[threadIdx.x + s];
            s2[threadIdx.x] += s2[threadIdx.x + s];
        }
        __syncthreads();
    }
    const float mu  = s1[0] * (1.0f / DIM);
    const float var = s2[0] * (1.0f / DIM) - mu * mu;
    const float inv = rsqrtf(var + LN_EPS);
    bf16_t* orow = out + (long long)row * DIM;
    for (int c = threadIdx.x; c < DIM; c += 256)
        orow[c] = (bf16_t)((xr[c] - mu) * inv * g[c] + b[c]);
}

// In-place softmax over rows of [BH*SEQ, SEQ], with multiplicative zero-mask
// on logits for rows 0..2 (faithful to reference).
__global__ __launch_bounds__(256) void softmax_kernel(float* __restrict__ w)
{
    __shared__ float sred[256];
    const long long r = blockIdx.x;
    const int n = (int)(r & (SEQ - 1));
    float* wr = w + r * SEQ;
    float vals[4];
    float mx = -3.4e38f;
#pragma unroll
    for (int t = 0; t < 4; ++t) {
        const int c = threadIdx.x + t * 256;
        float v = wr[c];
        const bool zero = (n == 0 && (c == 1 || c == 2)) ||
                          (n == 1 && (c == 0 || c == 2)) ||
                          (n == 2 && (c == 0 || c == 1));
        if (zero) v = 0.0f;
        vals[t] = v;
        mx = fmaxf(mx, v);
    }
    sred[threadIdx.x] = mx; __syncthreads();
    for (int s = 128; s > 0; s >>= 1) {
        if (threadIdx.x < s) sred[threadIdx.x] = fmaxf(sred[threadIdx.x], sred[threadIdx.x + s]);
        __syncthreads();
    }
    mx = sred[0];
    __syncthreads();
    float sum = 0.f;
#pragma unroll
    for (int t = 0; t < 4; ++t) { vals[t] = __expf(vals[t] - mx); sum += vals[t]; }
    sred[threadIdx.x] = sum; __syncthreads();
    for (int s = 128; s > 0; s >>= 1) {
        if (threadIdx.x < s) sred[threadIdx.x] += sred[threadIdx.x + s];
        __syncthreads();
    }
    const float inv = 1.0f / sred[0];
#pragma unroll
    for (int t = 0; t < 4; ++t) wr[threadIdx.x + t * 256] = vals[t] * inv;
}

// qkv [ROWS, 3*DIM] fp32 -> q_bf [B,H,N,D], k_bf [B,H,N,D], vT_bf [B,H,D,N] (bf16)
__global__ __launch_bounds__(256) void pack_qkv_kernel(
    const float* __restrict__ qkv, bf16_t* __restrict__ qb,
    bf16_t* __restrict__ kb, bf16_t* __restrict__ vt)
{
    const int idx = blockIdx.x * 256 + threadIdx.x;   // over BH*SEQ*HEAD_DIM
    const int d = idx & 63;
    int t = idx >> 6;
    const int n = t & (SEQ - 1);
    t >>= 10;
    const int h = t % HEADS;
    const int b = t / HEADS;
    const long long row = (long long)(b * SEQ + n) * (3 * DIM);
    const int c = h * HEAD_DIM + d;
    qb[idx] = (bf16_t)qkv[row + c];
    kb[idx] = (bf16_t)qkv[row + DIM + c];
    vt[(((long long)(b * HEADS + h) * HEAD_DIM + d) << 10) + n] =
        (bf16_t)qkv[row + 2 * DIM + c];
}

// o [B,H,N,D] fp32 -> attn_bf [B*N, H*D] bf16
__global__ __launch_bounds__(256) void permute_o_kernel(
    const float* __restrict__ o, bf16_t* __restrict__ out)
{
    const int idx = blockIdx.x * 256 + threadIdx.x;   // over ROWS*DIM
    const int col = idx % DIM;
    const int row = idx / DIM;
    const int d = col & 63, h = col >> 6;
    const int b = row >> 10, n = row & (SEQ - 1);
    out[idx] = (bf16_t)o[(((long long)(b * HEADS + h) << 10) + n) * HEAD_DIM + d];
}

// w [K,N] fp32 -> wt [N,K] bf16 (coalesced writes)
__global__ __launch_bounds__(256) void wt_kernel(
    const float* __restrict__ w, bf16_t* __restrict__ wt, int K, int N)
{
    const int idx = blockIdx.x * 256 + threadIdx.x;
    if (idx >= K * N) return;
    const int k = idx % K, n = idx / K;
    wt[idx] = (bf16_t)w[(long long)k * N + n];
}

extern "C" void kernel_launch(void* const* d_in, const int* in_sizes, int n_in,
                              void* d_out, int out_size, void* d_ws, size_t ws_size,
                              hipStream_t stream)
{
    (void)in_sizes; (void)n_in; (void)out_size; (void)ws_size;
    const float* x      = (const float*)d_in[0];
    const float* ln1_g  = (const float*)d_in[1];
    const float* ln1_b  = (const float*)d_in[2];
    const float* w_qkv  = (const float*)d_in[3];
    const float* w_proj = (const float*)d_in[4];
    const float* b_proj = (const float*)d_in[5];
    const float* ln2_g  = (const float*)d_in[6];
    const float* ln2_b  = (const float*)d_in[7];
    const float* w_fc1  = (const float*)d_in[8];
    const float* b_fc1  = (const float*)d_in[9];
    const float* w_fc2  = (const float*)d_in[10];
    const float* b_fc2  = (const float*)d_in[11];

    float* out_x = (float*)d_out;                       // [4,1024,768]
    float* out_w = out_x + (long long)ROWS * DIM;       // [4,12,1024,1024]

    char* ws = (char*)d_ws;
    size_t off = 0;
    auto alloc = [&](size_t bytes) -> char* {
        char* p = ws + off;
        off += (bytes + 255) & ~(size_t)255;
        return p;
    };

    float*  qkv    = (float*)alloc((size_t)ROWS * 3 * DIM * sizeof(float)); // 37.7MB
    bf16_t* m1     = (bf16_t*)qkv;  // alias: fc1 out [ROWS,HIDDEN] bf16 (25.2MB), qkv dead by then
    bf16_t* hb     = (bf16_t*)alloc((size_t)ROWS * DIM * 2);
    bf16_t* qb     = (bf16_t*)alloc((size_t)BH * SEQ * HEAD_DIM * 2);
    bf16_t* kb     = (bf16_t*)alloc((size_t)BH * SEQ * HEAD_DIM * 2);
    bf16_t* vt     = (bf16_t*)alloc((size_t)BH * SEQ * HEAD_DIM * 2);
    float*  obuf   = (float*)alloc((size_t)BH * SEQ * HEAD_DIM * sizeof(float));
    bf16_t* attnb  = (bf16_t*)alloc((size_t)ROWS * DIM * 2);
    float*  x1     = (float*)alloc((size_t)ROWS * DIM * sizeof(float));
    bf16_t* h2b    = (bf16_t*)alloc((size_t)ROWS * DIM * 2);
    bf16_t* wqkvT  = (bf16_t*)alloc((size_t)3 * DIM * DIM * 2);
    bf16_t* wprojT = (bf16_t*)alloc((size_t)DIM * DIM * 2);
    bf16_t* wfc1T  = (bf16_t*)alloc((size_t)HIDDEN * DIM * 2);
    bf16_t* wfc2T  = (bf16_t*)alloc((size_t)DIM * HIDDEN * 2);

    // Weight transpose+convert to bf16 B^T layout
    wt_kernel<<<(3 * DIM * DIM) / 256, 256, 0, stream>>>(w_qkv, wqkvT, DIM, 3 * DIM);
    wt_kernel<<<(DIM * DIM) / 256, 256, 0, stream>>>(w_proj, wprojT, DIM, DIM);
    wt_kernel<<<(HIDDEN * DIM) / 256, 256, 0, stream>>>(w_fc1, wfc1T, DIM, HIDDEN);
    wt_kernel<<<(DIM * HIDDEN) / 256, 256, 0, stream>>>(w_fc2, wfc2T, HIDDEN, DIM);

    // LN1 -> bf16
    ln_kernel<<<ROWS, 256, 0, stream>>>(x, ln1_g, ln1_b, hb);

    // qkv = h @ w_qkv   [4096,768] x [768,2304]
    gemm_wmma_kernel<EPI_PLAIN, bf16_t><<<dim3(36, 32, 1), 128, 0, stream>>>(
        hb, wqkvT, qkv, nullptr, nullptr, nullptr,
        ROWS, 3 * DIM, DIM, DIM, DIM, 3 * DIM, 0, 0, 0, 1.0f);

    pack_qkv_kernel<<<(BH * SEQ * HEAD_DIM) / 256, 256, 0, stream>>>(qkv, qb, kb, vt);

    // Logits = scale * Q K^T, batched over 48 (b,h), straight into d_out weights area
    gemm_wmma_kernel<EPI_SCALE, bf16_t><<<dim3(16, 8, BH), 128, 0, stream>>>(
        qb, kb, out_w, nullptr, nullptr, nullptr,
        SEQ, SEQ, HEAD_DIM, HEAD_DIM, HEAD_DIM, SEQ,
        (long long)SEQ * HEAD_DIM, (long long)SEQ * HEAD_DIM,
        (long long)SEQ * SEQ, ATT_SCALE);

    // mask + softmax in place
    softmax_kernel<<<BH * SEQ, 256, 0, stream>>>(out_w);

    // O = W @ V, A read as fp32 (converted in-register to bf16)
    gemm_wmma_kernel<EPI_PLAIN, float><<<dim3(1, 8, BH), 128, 0, stream>>>(
        out_w, vt, obuf, nullptr, nullptr, nullptr,
        SEQ, HEAD_DIM, SEQ, SEQ, SEQ, HEAD_DIM,
        (long long)SEQ * SEQ, (long long)SEQ * HEAD_DIM,
        (long long)SEQ * HEAD_DIM, 1.0f);

    permute_o_kernel<<<(ROWS * DIM) / 256, 256, 0, stream>>>(obuf, attnb);

    // x1 = x + O @ w_proj + b_proj
    gemm_wmma_kernel<EPI_BIAS_RESID, bf16_t><<<dim3(12, 32, 1), 128, 0, stream>>>(
        attnb, wprojT, x1, nullptr, b_proj, x,
        ROWS, DIM, DIM, DIM, DIM, DIM, 0, 0, 0, 1.0f);

    // LN2 -> bf16
    ln_kernel<<<ROWS, 256, 0, stream>>>(x1, ln2_g, ln2_b, h2b);

    // m1 = gelu(h2 @ w_fc1 + b_fc1), bf16 out
    gemm_wmma_kernel<EPI_BIAS_GELU_BF16, bf16_t><<<dim3(48, 32, 1), 128, 0, stream>>>(
        h2b, wfc1T, nullptr, m1, b_fc1, nullptr,
        ROWS, HIDDEN, DIM, DIM, DIM, HIDDEN, 0, 0, 0, 1.0f);

    // out_x = x1 + m1 @ w_fc2 + b_fc2
    gemm_wmma_kernel<EPI_BIAS_RESID, bf16_t><<<dim3(12, 32, 1), 128, 0, stream>>>(
        m1, wfc2T, out_x, nullptr, b_fc2, x1,
        ROWS, DIM, HIDDEN, HIDDEN, HIDDEN, DIM, 0, 0, 0, 1.0f);
}
